// Embed_82746839925254
// MI455X (gfx1250) — matrix operand order repaired and checked by reading, not compiled, not run on hardware
//
#include <hip/hip_runtime.h>
#include <hip/hip_bf16.h>

#define DEV __device__ __forceinline__

namespace {
constexpr int B_   = 128;
constexpr int L_   = 128;
constexpr int D_   = 128;
constexpr int H_   = 8;
constexpr int DK_  = 64;
constexpr int NL_  = 4;
constexpr int INF_ = 37;
constexpr int KP0_ = 64;             // padded K for layer-0 gconv (37 -> 64)
constexpr int N_   = B_ * L_;        // 16384
constexpr int E_   = N_ * 8;         // 131072
constexpr int QKVW_ = H_ * DK_;      // 512
constexpr int CTXW_ = DK_ * (H_ + 1);// 576
}

typedef __attribute__((ext_vector_type(16))) _Float16 v16h;
typedef __attribute__((ext_vector_type(8)))  _Float16 h8v;
typedef __attribute__((ext_vector_type(4)))  float    f4v;
typedef __attribute__((ext_vector_type(8)))  float    v8f;

// ---------------------------------------------------------------- wave utils
DEV float wsum(float x) {
#pragma unroll
  for (int off = 16; off; off >>= 1) x += __shfl_xor(x, off, 32);
  return x;
}
DEV float wmaxr(float x) {
#pragma unroll
  for (int off = 16; off; off >>= 1) x = fmaxf(x, __shfl_xor(x, off, 32));
  return x;
}

DEV v8f wmma32(v16h a, v16h b, v8f c) {
  return __builtin_amdgcn_wmma_f32_16x16x32_f16(false, a, false, b,
                                                (short)0, c, false, false);
}

// ---------------------------------------------------------------- fragments
// A fragment 16x32 (ISA 7.12.2): lane&15 = row M; lanes<16 hold K {0..7,16..23},
// lanes>=16 hold K {8..15,24..31}. K runs are contiguous -> vector loads.
DEV v16h fragA_h(const _Float16* A, int lda, int m0, int k0) {
  const int lane = threadIdx.x & 31;
  const _Float16* p =
      A + (long long)(m0 + (lane & 15)) * lda + k0 + ((lane >> 4) << 3);
  const h8v lo = *(const h8v*)p;
  const h8v hi = *(const h8v*)(p + 16);
  v16h f;
#pragma unroll
  for (int j = 0; j < 8; ++j) { f[j] = lo[j]; f[j + 8] = hi[j]; }
  return f;
}

DEV v16h fragA_f(const float* A, int lda, int m0, int k0) {
  const int lane = threadIdx.x & 31;
  const float* p =
      A + (long long)(m0 + (lane & 15)) * lda + k0 + ((lane >> 4) << 3);
  const f4v a0 = *(const f4v*)p;
  const f4v a1 = *(const f4v*)(p + 4);
  const f4v b0 = *(const f4v*)(p + 16);
  const f4v b1 = *(const f4v*)(p + 20);
  v16h f;
#pragma unroll
  for (int j = 0; j < 4; ++j) {
    f[j]      = (_Float16)a0[j];
    f[j + 4]  = (_Float16)a1[j];
    f[j + 8]  = (_Float16)b0[j];
    f[j + 12] = (_Float16)b1[j];
  }
  return f;
}

DEV v16h fragA(const _Float16* A, int lda, int m0, int k0) { return fragA_h(A, lda, m0, k0); }
DEV v16h fragA(const float*    A, int lda, int m0, int k0) { return fragA_f(A, lda, m0, k0); }

// B fragment 32x16 from B^T stored [N,K]: lane&15 = col N; lanes<16 hold
// K {0..7,16..23}, lanes>=16 hold K {8..15,24..31} (same order as A).
DEV v16h fragBt(const _Float16* Bt, int ldb, int n0, int k0) {
  const int lane = threadIdx.x & 31;
  const _Float16* p =
      Bt + (long long)(n0 + (lane & 15)) * ldb + k0 + ((lane >> 4) << 3);
  const h8v lo = *(const h8v*)p;
  const h8v hi = *(const h8v*)(p + 16);
  v16h f;
#pragma unroll
  for (int j = 0; j < 8; ++j) { f[j] = lo[j]; f[j + 8] = hi[j]; }
  return f;
}

// ---------------------------------------------------------------- GEMM kernels
// 4 waves / block, each wave owns a 16x16 tile along N. D = A * Bt^T (+bias)
// (+addend).  A: [M,K] row-major (f32 or f16); Bt: [Nc,K] f16.
template <typename TA, bool STORE_HALF>
__global__ __launch_bounds__(128)
void gemm_nt(const TA* A, int lda, const _Float16* Bt, int ldb,
             const float* __restrict__ bias, const float* __restrict__ addend,
             int ldadd, void* Dp, int ldd, int K) {
  const int wid = threadIdx.x >> 5;
  const int n0 = (blockIdx.x * 4 + wid) << 4;
  const int m0 = blockIdx.y << 4;
  v8f acc = {};
  for (int k0 = 0; k0 < K; k0 += 32)
    acc = wmma32(fragA(A, lda, m0, k0), fragBt(Bt, ldb, n0, k0), acc);
  const int lane = threadIdx.x & 31;
  const int n  = n0 + (lane & 15);
  const int mb = m0 + ((lane >> 4) << 3);
  const float bv = bias ? bias[n] : 0.0f;
#pragma unroll
  for (int r = 0; r < 8; ++r) {
    const int m = mb + r;
    float v = acc[r] + bv;
    if (addend) v += addend[(long long)m * ldadd + n];
    if (STORE_HALF) ((_Float16*)Dp)[(long long)m * ldd + n] = (_Float16)v;
    else            ((float*)Dp)[(long long)m * ldd + n]    = v;
  }
}

// Projection with per-molecule transposed store:
// out(m = b*128+l, n) -> Dt[(b*Ncols + n)*128 + l]   (produces V^T / v2^T)
__global__ __launch_bounds__(128)
void gemm_nt_tstore(const _Float16* A, int lda, const _Float16* Bt, int ldb,
                    _Float16* Dt, int Ncols, int K) {
  const int wid = threadIdx.x >> 5;
  const int n0 = (blockIdx.x * 4 + wid) << 4;
  const int m0 = blockIdx.y << 4;
  v8f acc = {};
  for (int k0 = 0; k0 < K; k0 += 32)
    acc = wmma32(fragA(A, lda, m0, k0), fragBt(Bt, ldb, n0, k0), acc);
  const int lane = threadIdx.x & 31;
  const int n  = n0 + (lane & 15);
  const int mb = m0 + ((lane >> 4) << 3);
#pragma unroll
  for (int r = 0; r < 8; ++r) {
    const int m = mb + r;
    const int b = m >> 7, l = m & 127;
    Dt[((long long)b * Ncols + n) * L_ + l] = (_Float16)acc[r];
  }
}

// scores[b,h,i,j] = (Q.K^T)/8, structural mask on heads 2..7.
__global__ __launch_bounds__(128)
void attn_scores_wmma(const _Float16* qbuf, const _Float16* kbuf,
                      const unsigned char* maskbits, float* attn) {
  const int wid = threadIdx.x >> 5;
  const int j0 = (blockIdx.x * 4 + wid) << 4;
  const int i0 = blockIdx.y << 4;
  const int bh = blockIdx.z;
  const int b  = bh >> 3;
  const int h  = bh & 7;
  const _Float16* Q  = qbuf + (long long)b * L_ * QKVW_ + h * DK_;
  const _Float16* Kt = kbuf + (long long)b * L_ * QKVW_ + h * DK_;
  v8f acc = {};
#pragma unroll
  for (int k0 = 0; k0 < DK_; k0 += 32)
    acc = wmma32(fragA(Q, QKVW_, i0, k0), fragBt(Kt, QKVW_, j0, k0), acc);
  const int lane = threadIdx.x & 31;
  const int j  = j0 + (lane & 15);
  const int ib = i0 + ((lane >> 4) << 3);
  const long long mrow = (long long)b * L_ * L_;
#pragma unroll
  for (int r = 0; r < 8; ++r) {
    const int i = ib + r;
    float s = acc[r] * 0.125f;                         // 1/sqrt(DK)
    if (h >= 2 && ((maskbits[mrow + (long long)i * L_ + j] >> (h - 2)) & 1))
      s = -1e9f;
    attn[((long long)bh * L_ + i) * L_ + j] = s;
  }
}

// ctx = attn @ V  (V supplied transposed: vt[(bh*64+dk)*128 + l]) -> ctxcat
__global__ __launch_bounds__(128)
void attn_ctx_wmma(const float* attn, const _Float16* vt, _Float16* ctxcat) {
  const int wid = threadIdx.x >> 5;
  const int n0 = wid << 4;                             // dk tile (DK=64 -> 4)
  const int i0 = blockIdx.y << 4;
  const int bh = blockIdx.z;
  const int b  = bh >> 3;
  const int h  = bh & 7;
  const float*    A  = attn + (long long)bh * L_ * L_;
  const _Float16* Bt = vt   + (long long)bh * DK_ * L_;
  v8f acc = {};
#pragma unroll
  for (int k0 = 0; k0 < L_; k0 += 32)
    acc = wmma32(fragA(A, L_, i0, k0), fragBt(Bt, L_, n0, k0), acc);
  const int lane = threadIdx.x & 31;
  const int n  = n0 + (lane & 15);
  const int ib = i0 + ((lane >> 4) << 3);
#pragma unroll
  for (int r = 0; r < 8; ++r) {
    const int i = ib + r;
    ctxcat[((long long)b * L_ + i) * CTXW_ + h * DK_ + n] = (_Float16)acc[r];
  }
}

// ctx2 = matrix @ v2 (v2 transposed: v2t[(b*64+dk)*128 + l]) -> ctxcat[:,512:]
__global__ __launch_bounds__(128)
void ctx2_wmma(const float* mat, const _Float16* v2t, _Float16* ctxcat) {
  const int wid = threadIdx.x >> 5;
  const int n0 = wid << 4;                             // dk tile
  const int i0 = blockIdx.y << 4;
  const int b  = blockIdx.z;
  const float*    A  = mat + (long long)b * L_ * L_;
  const _Float16* Bt = v2t + (long long)b * DK_ * L_;
  v8f acc = {};
#pragma unroll
  for (int k0 = 0; k0 < L_; k0 += 32)
    acc = wmma32(fragA(A, L_, i0, k0), fragBt(Bt, L_, n0, k0), acc);
  const int lane = threadIdx.x & 31;
  const int n  = n0 + (lane & 15);
  const int ib = i0 + ((lane >> 4) << 3);
#pragma unroll
  for (int r = 0; r < 8; ++r) {
    const int i = ib + r;
    ctxcat[((long long)b * L_ + i) * CTXW_ + QKVW_ + n] = (_Float16)acc[r];
  }
}

// ---------------------------------------------------------------- elementwise
__global__ void zero_f32(float* p, long long n) {
  long long i = (long long)blockIdx.x * blockDim.x + threadIdx.x;
  if (i < n) p[i] = 0.0f;
}

// pack W[K,N] (f32) -> Wt[N,Kpad] (f16), zero-padding K..Kpad-1.
__global__ void pack_wt(const float* W, _Float16* Wt, int K, int N, int Kpad) {
  long long idx = (long long)blockIdx.x * blockDim.x + threadIdx.x;
  if (idx >= (long long)N * Kpad) return;
  const int n = (int)(idx / Kpad);
  const int k = (int)(idx % Kpad);
  Wt[idx] = (k < K) ? (_Float16)W[(long long)k * N + n] : (_Float16)0.0f;
}

// x[N,37] -> xpad[N,64] zero-padded
__global__ void pad_x(const float* x, float* xp) {
  long long idx = (long long)blockIdx.x * blockDim.x + threadIdx.x;
  if (idx >= (long long)N_ * KP0_) return;
  const int r = (int)(idx >> 6);
  const int c = (int)(idx & 63);
  xp[idx] = (c < INF_) ? x[(long long)r * INF_ + c] : 0.0f;
}

__global__ void scatter_add(const float* h, int ldh, const int* edge,
                            float* agg, int ldagg, int F, int nE) {
  long long idx = (long long)blockIdx.x * blockDim.x + threadIdx.x;
  if (idx >= (long long)nE * F) return;
  const int e = (int)(idx / F);
  const int f = (int)(idx % F);
  const int s = edge[e];        // src
  const int d = edge[nE + e];   // dst
  atomicAdd(&agg[(long long)d * ldagg + f], h[(long long)s * ldh + f]);
}

// mask bitfield: bit m set == manu[m] masked (manu[m] -> head m+2)
__global__ void mask_init(const float* adj, const unsigned char* hete,
                          const int* ringm, const int* aromm, const int* alipm,
                          unsigned char* maskbits) {
  long long idx = (long long)blockIdx.x * blockDim.x + threadIdx.x;
  if (idx >= (long long)B_ * L_ * L_) return;
  const int b  = (int)(idx / (L_ * L_));
  const int ij = (int)(idx % (L_ * L_));
  const int i = ij / L_, j = ij % L_;
  const float a = adj[idx];
  const bool local = (a >= 0.8f) && (a != 0.825f);
  const bool coj   = a > 0.8f;
  const bool het   = hete[b * L_ + i] && hete[b * L_ + j];
  const int ri = ringm[b * L_ + i], rj = ringm[b * L_ + j];
  const int ai = aromm[b * L_ + i], aj = aromm[b * L_ + j];
  const int li = alipm[b * L_ + i], lj = alipm[b * L_ + j];
  const bool segr = (ri == rj) && (ri > 0);
  const bool sega = (ai == aj) && (ai > 0);
  const bool segl = (li == lj) && (li > 0);
  unsigned char m = 0;
  m |= (!local) ? 1u  : 0u;
  m |= (!coj)   ? 2u  : 0u;
  m |= (!het)   ? 4u  : 0u;
  m |= (!segr)  ? 8u  : 0u;
  m |= (!sega)  ? 16u : 0u;
  m |= (!segl)  ? 32u : 0u;
  maskbits[idx] = m;
}

__global__ void matrix_init(const float* adj, const float* dis, float* mat) {
  long long idx = (long long)blockIdx.x * blockDim.x + threadIdx.x;
  if (idx >= (long long)B_ * L_ * L_) return;
  const float a = adj[idx];
  const float d = dis[idx];
  const float dis_ = (d == 0.0f) ? 0.0f : rsqrtf(d);
  mat[idx] = (a == 0.0f) ? dis_ : a;
}

// h -> relu(LN(h)); writes f32 (xb) and f16 (xh). One wave per 128-row.
__global__ void relu_ln(const float* hbuf, float* xb, _Float16* xh, int nrows) {
  const int row = blockIdx.x * (blockDim.x >> 5) + (threadIdx.x >> 5);
  if (row >= nrows) return;
  const int lane = threadIdx.x & 31;
  const float* p = hbuf + (long long)row * D_;
  float v[4];
  float s = 0.0f;
#pragma unroll
  for (int t = 0; t < 4; ++t) { v[t] = p[lane + 32 * t]; s += v[t]; }
  const float mean = wsum(s) * (1.0f / D_);
  float q = 0.0f;
#pragma unroll
  for (int t = 0; t < 4; ++t) { const float d = v[t] - mean; q += d * d; }
  const float inv = rsqrtf(wsum(q) * (1.0f / D_) + 1e-5f);
#pragma unroll
  for (int t = 0; t < 4; ++t) {
    float y = fmaxf((v[t] - mean) * inv, 0.0f);
    xb[(long long)row * D_ + lane + 32 * t] = y;
    xh[(long long)row * D_ + lane + 32 * t] = (_Float16)y;
  }
}

__global__ void softmax_rows(float* attn, int nrows) {
  const int row = blockIdx.x * (blockDim.x >> 5) + (threadIdx.x >> 5);
  if (row >= nrows) return;
  const int lane = threadIdx.x & 31;
  float* p = attn + (long long)row * L_;
  float v[4];
  float mx = -3.4e38f;
#pragma unroll
  for (int t = 0; t < 4; ++t) { v[t] = p[lane + 32 * t]; mx = fmaxf(mx, v[t]); }
  mx = wmaxr(mx);
  float s = 0.0f;
#pragma unroll
  for (int t = 0; t < 4; ++t) { v[t] = __expf(v[t] - mx); s += v[t]; }
  const float inv = 1.0f / wsum(s);
#pragma unroll
  for (int t = 0; t < 4; ++t) p[lane + 32 * t] = v[t] * inv;
}

// out = LN(sigmoid(LN(fcout)) + xb_in); writes f32 and f16 copies.
__global__ void enc_post(const float* fcout, const float* xb_in,
                         float* xb_out, _Float16* xh, int nrows) {
  const int row = blockIdx.x * (blockDim.x >> 5) + (threadIdx.x >> 5);
  if (row >= nrows) return;
  const int lane = threadIdx.x & 31;
  const float* p = fcout + (long long)row * D_;
  const float* q = xb_in + (long long)row * D_;
  float v[4];
  float s = 0.0f;
#pragma unroll
  for (int t = 0; t < 4; ++t) { v[t] = p[lane + 32 * t]; s += v[t]; }
  const float m1 = wsum(s) * (1.0f / D_);
  float qq = 0.0f;
#pragma unroll
  for (int t = 0; t < 4; ++t) { const float d = v[t] - m1; qq += d * d; }
  const float i1 = rsqrtf(wsum(qq) * (1.0f / D_) + 1e-5f);
  float u[4];
  float s2 = 0.0f;
#pragma unroll
  for (int t = 0; t < 4; ++t) {
    const float y = (v[t] - m1) * i1;
    u[t] = 1.0f / (1.0f + __expf(-y)) + q[lane + 32 * t];
    s2 += u[t];
  }
  const float m2 = wsum(s2) * (1.0f / D_);
  float q2 = 0.0f;
#pragma unroll
  for (int t = 0; t < 4; ++t) { const float d = u[t] - m2; q2 += d * d; }
  const float i2 = rsqrtf(wsum(q2) * (1.0f / D_) + 1e-5f);
#pragma unroll
  for (int t = 0; t < 4; ++t) {
    const float y = (u[t] - m2) * i2;
    xb_out[(long long)row * D_ + lane + 32 * t] = y;
    xh[(long long)row * D_ + lane + 32 * t] = (_Float16)y;
  }
}

// gating MLP over (matrix, attn) channels (9): LN -> 9x6 relu -> 6x1 sigmoid
__global__ void gate_update(const float* attn, const float* mat_in,
                            float* mat_out,
                            const float* lng, const float* lnb,
                            const float* W1, const float* b1,
                            const float* W2, const float* b2) {
  long long idx = (long long)blockIdx.x * blockDim.x + threadIdx.x;
  if (idx >= (long long)B_ * L_ * L_) return;
  const int b  = (int)(idx / (L_ * L_));
  const int ij = (int)(idx % (L_ * L_));
  float v[9];
  v[0] = mat_in[idx];
#pragma unroll
  for (int h = 0; h < H_; ++h)
    v[1 + h] = attn[((long long)(b * H_ + h)) * L_ * L_ + ij];
  float s = 0.0f;
#pragma unroll
  for (int c = 0; c < 9; ++c) s += v[c];
  const float mean = s * (1.0f / 9.0f);
  float q = 0.0f;
#pragma unroll
  for (int c = 0; c < 9; ++c) { const float d = v[c] - mean; q += d * d; }
  const float inv = rsqrtf(q * (1.0f / 9.0f) + 1e-5f);
  float t[9];
#pragma unroll
  for (int c = 0; c < 9; ++c) t[c] = (v[c] - mean) * inv * lng[c] + lnb[c];
  float g = b2[0];
#pragma unroll
  for (int k = 0; k < 6; ++k) {
    float h1 = b1[k];
#pragma unroll
    for (int c = 0; c < 9; ++c) h1 += t[c] * W1[c * 6 + k];
    g += fmaxf(h1, 0.0f) * W2[k];
  }
  g = 1.0f / (1.0f + __expf(-g));
  mat_out[idx] = v[0] * g;
}

// ---------------------------------------------------------------- host driver
extern "C" void kernel_launch(void* const* d_in, const int* in_sizes, int n_in,
                              void* d_out, int out_size, void* d_ws, size_t ws_size,
                              hipStream_t stream) {
  (void)in_sizes; (void)n_in; (void)out_size; (void)ws_size;
  const float* x    = (const float*)d_in[0];
  const int*   edge = (const int*)d_in[1];
  const float* adj  = (const float*)d_in[5];
  const float* dis  = (const float*)d_in[6];
  const int* ringm  = (const int*)d_in[7];
  const int* aromm  = (const int*)d_in[8];
  const int* alipm  = (const int*)d_in[9];
  const unsigned char* hete = (const unsigned char*)d_in[10];
  const float* g0_Wrel  = (const float*)d_in[11];
  const float* g0_brel  = (const float*)d_in[12];
  const float* g0_Wroot = (const float*)d_in[13];
  const float* g_Wrel   = (const float*)d_in[14];
  const float* g_brel   = (const float*)d_in[15];
  const float* g_Wroot  = (const float*)d_in[16];
  const float* WQ  = (const float*)d_in[17];
  const float* WK  = (const float*)d_in[18];
  const float* WV  = (const float*)d_in[19];
  const float* WV2 = (const float*)d_in[20];
  const float* fcW = (const float*)d_in[21];
  const float* fu_lng = (const float*)d_in[22];
  const float* fu_lnb = (const float*)d_in[23];
  const float* fu_W1  = (const float*)d_in[24];
  const float* fu_b1  = (const float*)d_in[25];
  const float* fu_W2  = (const float*)d_in[26];
  const float* fu_b2  = (const float*)d_in[27];

  char* w = (char*)d_ws;
  auto alloc = [&](size_t bytes) -> void* {
    void* p = (void*)w;
    w += (bytes + 255) & ~(size_t)255;
    return p;
  };
  float*    agg    = (float*)alloc(sizeof(float) * (size_t)N_ * D_);
  float*    hbuf   = (float*)alloc(sizeof(float) * (size_t)N_ * D_);
  float*    xb     = (float*)alloc(sizeof(float) * (size_t)N_ * D_);
  float*    xpad   = (float*)alloc(sizeof(float) * (size_t)N_ * KP0_);
  _Float16* xh     = (_Float16*)alloc(sizeof(_Float16) * (size_t)N_ * D_);
  _Float16* qbuf   = (_Float16*)alloc(sizeof(_Float16) * (size_t)N_ * QKVW_);
  _Float16* kbuf   = (_Float16*)alloc(sizeof(_Float16) * (size_t)N_ * QKVW_);
  _Float16* vt     = (_Float16*)alloc(sizeof(_Float16) * (size_t)N_ * QKVW_);
  _Float16* v2t    = (_Float16*)alloc(sizeof(_Float16) * (size_t)N_ * DK_);
  float*    attn   = (float*)alloc(sizeof(float) * (size_t)B_ * H_ * L_ * L_);
  _Float16* ctxcat = (_Float16*)alloc(sizeof(_Float16) * (size_t)N_ * CTXW_);
  float*    fcout  = (float*)alloc(sizeof(float) * (size_t)N_ * D_);
  float*    matA   = (float*)alloc(sizeof(float) * (size_t)B_ * L_ * L_);
  float*    matB   = (float*)alloc(sizeof(float) * (size_t)B_ * L_ * L_);
  unsigned char* maskbits =
      (unsigned char*)alloc(sizeof(unsigned char) * (size_t)B_ * L_ * L_);
  // packed f16-transposed weights (reused per layer)
  _Float16* wrel_t  = (_Float16*)alloc(sizeof(_Float16) * (size_t)D_ * D_);
  _Float16* wroot_t = (_Float16*)alloc(sizeof(_Float16) * (size_t)D_ * D_);
  _Float16* wq_t    = (_Float16*)alloc(sizeof(_Float16) * (size_t)QKVW_ * D_);
  _Float16* wk_t    = (_Float16*)alloc(sizeof(_Float16) * (size_t)QKVW_ * D_);
  _Float16* wv_t    = (_Float16*)alloc(sizeof(_Float16) * (size_t)QKVW_ * D_);
  _Float16* wv2_t   = (_Float16*)alloc(sizeof(_Float16) * (size_t)DK_ * D_);
  _Float16* fc_t    = (_Float16*)alloc(sizeof(_Float16) * (size_t)D_ * CTXW_);

  const long long BLL = (long long)B_ * L_ * L_;
  mask_init<<<(unsigned)((BLL + 255) / 256), 256, 0, stream>>>(
      adj, hete, ringm, aromm, alipm, maskbits);
  matrix_init<<<(unsigned)((BLL + 255) / 256), 256, 0, stream>>>(adj, dis, matA);
  pad_x<<<(N_ * KP0_) / 256, 256, 0, stream>>>(x, xpad);

  auto pack = [&](const float* W, _Float16* Wt, int K, int N, int Kpad) {
    const long long cnt = (long long)N * Kpad;
    pack_wt<<<(unsigned)((cnt + 255) / 256), 256, 0, stream>>>(W, Wt, K, N, Kpad);
  };

  float* mat_in  = matA;
  float* mat_out = matB;

  for (int i = 0; i < NL_; ++i) {
    const int F    = (i == 0) ? INF_ : D_;
    const int Kpad = (i == 0) ? KP0_ : D_;
    const float* hin   = (i == 0) ? xpad : xb;          // row stride = Kpad
    const float* Wrel  = (i == 0) ? g0_Wrel  : g_Wrel  + (long long)(i - 1) * D_ * D_;
    const float* brel  = (i == 0) ? g0_brel  : g_brel  + (long long)(i - 1) * D_;
    const float* Wroot = (i == 0) ? g0_Wroot : g_Wroot + (long long)(i - 1) * D_ * D_;

    // pack this layer's weights to f16 transposed [N,Kpad]
    pack(Wrel,  wrel_t,  F, D_, Kpad);
    pack(Wroot, wroot_t, F, D_, Kpad);
    pack(WQ  + (long long)i * D_ * QKVW_, wq_t,  D_, QKVW_, D_);
    pack(WK  + (long long)i * D_ * QKVW_, wk_t,  D_, QKVW_, D_);
    pack(WV  + (long long)i * D_ * QKVW_, wv_t,  D_, QKVW_, D_);
    pack(WV2 + (long long)i * D_ * DK_,   wv2_t, D_, DK_,   D_);
    pack(fcW + (long long)i * CTXW_ * D_, fc_t,  CTXW_, D_, CTXW_);

    // --- GraphConv: agg = segment_sum(h[src], dst); h = agg@Wrel+brel+h@Wroot
    const long long nagg = (long long)N_ * Kpad;        // zero incl. K padding
    zero_f32<<<(unsigned)((nagg + 255) / 256), 256, 0, stream>>>(agg, nagg);
    const long long nsc = (long long)E_ * F;
    scatter_add<<<(unsigned)((nsc + 255) / 256), 256, 0, stream>>>(
        hin, Kpad, edge, agg, Kpad, F, E_);
    gemm_nt<float, false><<<dim3(D_ / 64, N_ / 16), 128, 0, stream>>>(
        agg, Kpad, wrel_t, Kpad, brel, nullptr, 0, hbuf, D_, Kpad);
    gemm_nt<float, false><<<dim3(D_ / 64, N_ / 16), 128, 0, stream>>>(
        hin, Kpad, wroot_t, Kpad, nullptr, hbuf, D_, hbuf, D_, Kpad);
    relu_ln<<<N_ / 8, 256, 0, stream>>>(hbuf, xb, xh, N_);

    // --- Encoder layer i ---
    gemm_nt<_Float16, true><<<dim3(QKVW_ / 64, N_ / 16), 128, 0, stream>>>(
        xh, D_, wq_t, D_, nullptr, nullptr, 0, qbuf, QKVW_, D_);
    gemm_nt<_Float16, true><<<dim3(QKVW_ / 64, N_ / 16), 128, 0, stream>>>(
        xh, D_, wk_t, D_, nullptr, nullptr, 0, kbuf, QKVW_, D_);
    gemm_nt_tstore<<<dim3(QKVW_ / 64, N_ / 16), 128, 0, stream>>>(
        xh, D_, wv_t, D_, vt, QKVW_, D_);
    gemm_nt_tstore<<<dim3(DK_ / 64, N_ / 16), 128, 0, stream>>>(
        xh, D_, wv2_t, D_, v2t, DK_, D_);

    attn_scores_wmma<<<dim3(L_ / 64, L_ / 16, B_ * H_), 128, 0, stream>>>(
        qbuf, kbuf, maskbits, attn);
    softmax_rows<<<(B_ * H_ * L_) / 8, 256, 0, stream>>>(attn, B_ * H_ * L_);

    gate_update<<<(unsigned)((BLL + 255) / 256), 256, 0, stream>>>(
        attn, mat_in, mat_out,
        fu_lng + (long long)i * (H_ + 1), fu_lnb + (long long)i * (H_ + 1),
        fu_W1 + (long long)i * (H_ + 1) * 6, fu_b1 + (long long)i * 6,
        fu_W2 + (long long)i * 6, fu_b2 + (long long)i);

    ctx2_wmma<<<dim3(1, L_ / 16, B_), 128, 0, stream>>>(mat_in, v2t, ctxcat);
    attn_ctx_wmma<<<dim3(1, L_ / 16, B_ * H_), 128, 0, stream>>>(attn, vt, ctxcat);

    gemm_nt<_Float16, false><<<dim3(D_ / 64, N_ / 16), 128, 0, stream>>>(
        ctxcat, CTXW_, fc_t, CTXW_, nullptr, nullptr, 0, fcout, D_, CTXW_);

    float* xb_next = (i == NL_ - 1) ? (float*)d_out : xb;
    enc_post<<<N_ / 8, 256, 0, stream>>>(fcout, xb, xb_next, xh, N_);

    float* tmp = mat_in; mat_in = mat_out; mat_out = tmp;
  }
}
